// EncoderRLLM_32658931319558
// MI455X (gfx1250) — matrix operand order repaired and checked by reading, not compiled, not run on hardware
//
#include <hip/hip_runtime.h>
#include <hip/hip_bf16.h>
#include <math.h>

#define BATCH 8
#define SEQ   1024
#define DIM   768
#define NTOK  (BATCH*SEQ)

typedef __attribute__((ext_vector_type(16))) __bf16        v16bf;
typedef __attribute__((ext_vector_type(8)))  float         v8f;
typedef __attribute__((ext_vector_type(4)))  unsigned int  v4u;
typedef __attribute__((ext_vector_type(2)))  unsigned int  v2u;
typedef __attribute__((ext_vector_type(4)))  float         v4f;

__device__ __forceinline__ unsigned short f2bf(float f) {
    unsigned int x = __float_as_uint(f);
    x += 0x7FFFu + ((x >> 16) & 1u);
    return (unsigned short)(x >> 16);
}
__device__ __forceinline__ float bf2f(unsigned short u) {
    return __uint_as_float(((unsigned int)u) << 16);
}
__device__ __forceinline__ v8f zero8() {
    v8f z = {0.f,0.f,0.f,0.f,0.f,0.f,0.f,0.f};
    return z;
}

// A fragment (16x32 bf16): rowbase = base + (row_of_lane)*ld + k0.
// Lane l holds row l&15; 16 contiguous K values starting at (l>>4)*16.
__device__ __forceinline__ v16bf frag_a(const unsigned short* rowbase, int lane) {
    const unsigned short* p = rowbase + ((lane >> 4) << 4);
    union { v16bf v; v4u q[2]; } u;
    u.q[0] = *reinterpret_cast<const v4u*>(p);
    u.q[1] = *reinterpret_cast<const v4u*>(p + 8);
    return u.v;
}
// B fragment (32x16 bf16) from column-major-stored B (column contiguous over K):
// colbase = base + (col_of_lane)*ld + k0. Lane l holds col l&15; 16 contiguous K
// values starting at (l>>4)*16.
__device__ __forceinline__ v16bf frag_b(const unsigned short* colbase, int lane) {
    const unsigned short* p = colbase + ((lane >> 4) << 4);
    union { v16bf v; v4u q[2]; } u;
    u.q[0] = *reinterpret_cast<const v4u*>(p);
    u.q[1] = *reinterpret_cast<const v4u*>(p + 8);
    return u.v;
}

#define WMMA_BF16(A, B, C) \
    __builtin_amdgcn_wmma_f32_16x16x32_bf16(false, (A), false, (B), (short)0, (C), false, false)

// ---------------------------------------------------------------- weights: f32 [K,N] -> bf16 [N,K]
__global__ __launch_bounds__(256)
void transpose_w_kernel(const float* __restrict__ W, unsigned short* __restrict__ WT) {
    int gid = blockIdx.x * 256 + threadIdx.x;        // 768*768 threads
    int n = gid / DIM, k = gid % DIM;
    WT[(size_t)n * DIM + k] = f2bf(W[(size_t)k * DIM + n]);
}

// ---------------------------------------------------------------- x = emb[idx] + pos  -> bf16
__global__ __launch_bounds__(256)
void embed_kernel(const int* __restrict__ idx, const float* __restrict__ emb,
                  const float* __restrict__ pos, unsigned short* __restrict__ xbf) {
    int gid = blockIdx.x * 256 + threadIdx.x;        // NTOK * 192 threads, 4 floats each
    int t = gid / 192;
    int j = (gid % 192) * 4;
    int s = t & (SEQ - 1);
    v4f e = *reinterpret_cast<const v4f*>(emb + (size_t)idx[t] * DIM + j);
    v4f p = *reinterpret_cast<const v4f*>(pos + (size_t)s * DIM + j);
    v2u o;
    o[0] = (unsigned int)f2bf(e[0] + p[0]) | ((unsigned int)f2bf(e[1] + p[1]) << 16);
    o[1] = (unsigned int)f2bf(e[2] + p[2]) | ((unsigned int)f2bf(e[3] + p[3]) << 16);
    *reinterpret_cast<v2u*>(xbf + (size_t)t * DIM + j) = o;
}

// ---------------------------------------------------------------- GEMM: out = epilogue(A[M,768] @ WT[N,768]^T)
// mode 0: bf16 row-major out; mode 1: bf16 transposed-per-batch out (v^T [B,D,S]);
// mode 2: bias + exact GELU -> bf16; mode 3: bias + exact GELU -> f32
__global__ __launch_bounds__(128)
void gemm_wmma_kernel(const unsigned short* __restrict__ A,
                      const unsigned short* __restrict__ WT,
                      const float* __restrict__ bias,
                      void* __restrict__ out, int mode) {
    const int lane = threadIdx.x & 31;
    const int wave = threadIdx.x >> 5;
    const int m0 = blockIdx.x * 128 + (wave >> 1) * 64;
    const int n0 = blockIdx.y * 128 + (wave & 1) * 64;
    const int r  = lane & 15;
    const int hf = lane >> 4;

    v8f acc[4][4];
    #pragma unroll
    for (int i = 0; i < 4; i++)
        #pragma unroll
        for (int j = 0; j < 4; j++) acc[i][j] = zero8();

    for (int k0 = 0; k0 < DIM; k0 += 32) {
        v16bf af[4], bfm[4];
        #pragma unroll
        for (int i = 0; i < 4; i++)
            af[i] = frag_a(A + (size_t)(m0 + i * 16 + r) * DIM + k0, lane);
        #pragma unroll
        for (int j = 0; j < 4; j++)
            bfm[j] = frag_b(WT + (size_t)(n0 + j * 16 + r) * DIM + k0, lane);
        #pragma unroll
        for (int i = 0; i < 4; i++)
            #pragma unroll
            for (int j = 0; j < 4; j++)
                acc[i][j] = WMMA_BF16(af[i], bfm[j], acc[i][j]);
    }

    #pragma unroll
    for (int i = 0; i < 4; i++) {
        #pragma unroll
        for (int j = 0; j < 4; j++) {
            #pragma unroll
            for (int rr = 0; rr < 8; rr++) {
                int m = m0 + i * 16 + rr + hf * 8;
                int n = n0 + j * 16 + r;
                float v = acc[i][j][rr];
                if (mode >= 2) {
                    v += bias[n];
                    v = 0.5f * v * (1.0f + erff(v * 0.70710678118654752f));  // exact GELU
                }
                if (mode == 1) {            // v^T: [B, D, S]
                    int b = m >> 10, s = m & (SEQ - 1);
                    ((unsigned short*)out)[(size_t)b * DIM * SEQ + (size_t)n * SEQ + s] = f2bf(v);
                } else if (mode == 3) {
                    ((float*)out)[(size_t)m * DIM + n] = v;
                } else {
                    ((unsigned short*)out)[(size_t)m * DIM + n] = f2bf(v);
                }
            }
        }
    }
}

// ---------------------------------------------------------------- attention
// One wave handles 16 query rows (full 1024 keys).  4 waves per block.
// Per-wave LDS slice: probs bf16 [16][1024] (32KB) + rowmax[16] + rowsum[16].
#define ATT_SLICE (32768 + 128)
__global__ __launch_bounds__(128)
void attention_kernel(const unsigned short* __restrict__ qbf,   // [B*S, D] bf16
                      const unsigned short* __restrict__ kbf,   // [B*S, D] bf16
                      const unsigned short* __restrict__ vT,    // [B, D, S] bf16
                      const int* __restrict__ amask,            // [B, S]
                      float* __restrict__ aout) {               // [B*S, D] f32
    extern __shared__ unsigned char smem[];
    const int lane = threadIdx.x & 31;
    const int wave = threadIdx.x >> 5;
    unsigned short* probs = (unsigned short*)(smem + (size_t)wave * ATT_SLICE);
    float* rowmax = (float*)(smem + (size_t)wave * ATT_SLICE + 32768);
    float* rowsum = rowmax + 16;

    const int b  = blockIdx.y;
    const int q0 = blockIdx.x * 64 + wave * 16;
    const unsigned short* qb = qbf + (size_t)(b * SEQ + q0) * DIM;
    const unsigned short* kb = kbf + (size_t)b * SEQ * DIM;
    const unsigned short* vb = vT  + (size_t)b * DIM * SEQ;
    const int* mb = amask + b * SEQ;

    const int r  = lane & 15;
    const int hf = lane >> 4;
    const float scale = 0.03608439182435161f;   // 1/sqrt(768)

    // q rows resident in registers: 24 A-fragments (16 x 768 bf16)
    v16bf qf[24];
    #pragma unroll
    for (int kk = 0; kk < 24; kk++)
        qf[kk] = frag_a(qb + (size_t)r * DIM + kk * 32, lane);

    // ---- pass A: scores = mask(scale * q k^T), track running row max ----
    float rmax[8];
    #pragma unroll
    for (int rr = 0; rr < 8; rr++) rmax[rr] = -3.0e38f;

    for (int nb = 0; nb < 64; nb++) {
        v8f acc = zero8();
        #pragma unroll
        for (int kk = 0; kk < 24; kk++) {
            v16bf bfr = frag_b(kb + (size_t)(nb * 16 + r) * DIM + kk * 32, lane);
            acc = WMMA_BF16(qf[kk], bfr, acc);
        }
        int col = nb * 16 + r;
        float mval = (mb[col] == 0) ? -1.0e30f : 0.0f;
        #pragma unroll
        for (int rr = 0; rr < 8; rr++) {
            float s = acc[rr] * scale + mval;
            rmax[rr] = fmaxf(rmax[rr], s);
            probs[(rr + hf * 8) * SEQ + col] = f2bf(s);
        }
    }
    // reduce row max across the 16 lanes of each half, publish to LDS
    #pragma unroll
    for (int rr = 0; rr < 8; rr++) {
        float v = rmax[rr];
        v = fmaxf(v, __shfl_xor(v, 1, 32));
        v = fmaxf(v, __shfl_xor(v, 2, 32));
        v = fmaxf(v, __shfl_xor(v, 4, 32));
        v = fmaxf(v, __shfl_xor(v, 8, 32));
        if (r == 0) rowmax[hf * 8 + rr] = v;
    }
    __syncthreads();

    // ---- pass B: in-place exp(score - max) and row sums ----
    {
        float rm = rowmax[r];
        float sum = 0.f;
        unsigned short* prow = probs + (size_t)r * SEQ + hf * 512;
        for (int kk = 0; kk < 512; kk += 8) {
            union { v4u q; unsigned short h[8]; } u;
            u.q = *reinterpret_cast<const v4u*>(prow + kk);
            #pragma unroll
            for (int e = 0; e < 8; e++) {
                float ev = __expf(bf2f(u.h[e]) - rm);
                sum += ev;
                u.h[e] = f2bf(ev);
            }
            *reinterpret_cast<v4u*>(prow + kk) = u.q;
        }
        sum += __shfl_xor(sum, 16, 32);
        if (hf == 0) rowsum[r] = sum;
    }
    __syncthreads();

    // ---- pass C: out = (P @ V) / rowsum ----
    float rinv[8];
    #pragma unroll
    for (int rr = 0; rr < 8; rr++) rinv[rr] = 1.0f / rowsum[rr + hf * 8];

    for (int dbt = 0; dbt < 12; dbt++) {          // 12 * 4 * 16 = 768 output cols
        v8f acc[4];
        #pragma unroll
        for (int j = 0; j < 4; j++) acc[j] = zero8();
        for (int kk = 0; kk < 32; kk++) {         // 32 * 32 = 1024 keys
            v16bf af = frag_a(probs + (size_t)r * SEQ + kk * 32, lane);
            #pragma unroll
            for (int j = 0; j < 4; j++) {
                v16bf bfr = frag_b(vb + (size_t)((dbt * 4 + j) * 16 + r) * SEQ + kk * 32, lane);
                acc[j] = WMMA_BF16(af, bfr, acc[j]);
            }
        }
        #pragma unroll
        for (int j = 0; j < 4; j++)
            #pragma unroll
            for (int rr = 0; rr < 8; rr++) {
                int m = b * SEQ + q0 + rr + hf * 8;
                aout[(size_t)m * DIM + (dbt * 4 + j) * 16 + r] = acc[j][rr] * rinv[rr];
            }
    }
}

// ---------------------------------------------------------------- layernorm (one block per row)
// mode 0 -> bf16 out, mode 1 -> f32 out
__global__ __launch_bounds__(256)
void layernorm_kernel(const float* __restrict__ x, const float* __restrict__ g,
                      const float* __restrict__ be, void* __restrict__ out, int mode) {
    __shared__ float red[256];
    const int row = blockIdx.x;
    const int tid = threadIdx.x;
    const float* xr = x + (size_t)row * DIM;
    float v0 = xr[tid], v1 = xr[tid + 256], v2 = xr[tid + 512];

    red[tid] = v0 + v1 + v2;
    __syncthreads();
    for (int off = 128; off > 0; off >>= 1) {
        if (tid < off) red[tid] += red[tid + off];
        __syncthreads();
    }
    float mu = red[0] * (1.0f / 768.0f);
    __syncthreads();

    float d0 = v0 - mu, d1 = v1 - mu, d2 = v2 - mu;
    red[tid] = d0 * d0 + d1 * d1 + d2 * d2;
    __syncthreads();
    for (int off = 128; off > 0; off >>= 1) {
        if (tid < off) red[tid] += red[tid + off];
        __syncthreads();
    }
    float rstd = rsqrtf(red[0] * (1.0f / 768.0f) + 1e-5f);

    float y0 = d0 * rstd * g[tid]       + be[tid];
    float y1 = d1 * rstd * g[tid + 256] + be[tid + 256];
    float y2 = d2 * rstd * g[tid + 512] + be[tid + 512];
    if (mode == 0) {
        unsigned short* o = (unsigned short*)out + (size_t)row * DIM;
        o[tid] = f2bf(y0); o[tid + 256] = f2bf(y1); o[tid + 512] = f2bf(y2);
    } else {
        float* o = (float*)out + (size_t)row * DIM;
        o[tid] = y0; o[tid + 256] = y1; o[tid + 512] = y2;
    }
}

// ---------------------------------------------------------------- launch
extern "C" void kernel_launch(void* const* d_in, const int* in_sizes, int n_in,
                              void* d_out, int out_size, void* d_ws, size_t ws_size,
                              hipStream_t stream) {
    (void)in_sizes; (void)n_in; (void)out_size; (void)ws_size;
    const int*   idx   = (const int*)  d_in[0];
    const int*   amask = (const int*)  d_in[1];
    const float* emb   = (const float*)d_in[2];
    const float* pos   = (const float*)d_in[3];
    const float* Wq    = (const float*)d_in[4];
    const float* Wk    = (const float*)d_in[5];
    const float* Wv    = (const float*)d_in[6];
    const float* ln1g  = (const float*)d_in[7];
    const float* ln1b  = (const float*)d_in[8];
    const float* W1    = (const float*)d_in[9];
    const float* b1    = (const float*)d_in[10];
    const float* W2    = (const float*)d_in[11];
    const float* b2    = (const float*)d_in[12];
    const float* ln2g  = (const float*)d_in[13];
    const float* ln2b  = (const float*)d_in[14];

    unsigned char* ws = (unsigned char*)d_ws;
    size_t off = 0;
    auto alloc = [&](size_t bytes) {
        void* p = ws + off;
        off += (bytes + 255) & ~(size_t)255;
        return p;
    };
    unsigned short* WqT  = (unsigned short*)alloc((size_t)DIM * DIM * 2);
    unsigned short* WkT  = (unsigned short*)alloc((size_t)DIM * DIM * 2);
    unsigned short* WvT  = (unsigned short*)alloc((size_t)DIM * DIM * 2);
    unsigned short* W1T  = (unsigned short*)alloc((size_t)DIM * DIM * 2);
    unsigned short* W2T  = (unsigned short*)alloc((size_t)DIM * DIM * 2);
    unsigned short* xbf  = (unsigned short*)alloc((size_t)NTOK * DIM * 2);
    unsigned short* qbf  = (unsigned short*)alloc((size_t)NTOK * DIM * 2);
    unsigned short* kbf  = (unsigned short*)alloc((size_t)NTOK * DIM * 2);
    unsigned short* vTb  = (unsigned short*)alloc((size_t)NTOK * DIM * 2);
    float*          af32 = (float*)        alloc((size_t)NTOK * DIM * 4);  // attn out, reused for h2
    unsigned short* ln1o = (unsigned short*)alloc((size_t)NTOK * DIM * 2);
    unsigned short* h1o  = (unsigned short*)alloc((size_t)NTOK * DIM * 2);

    // bf16-transposed weights
    transpose_w_kernel<<<DIM * DIM / 256, 256, 0, stream>>>(Wq, WqT);
    transpose_w_kernel<<<DIM * DIM / 256, 256, 0, stream>>>(Wk, WkT);
    transpose_w_kernel<<<DIM * DIM / 256, 256, 0, stream>>>(Wv, WvT);
    transpose_w_kernel<<<DIM * DIM / 256, 256, 0, stream>>>(W1, W1T);
    transpose_w_kernel<<<DIM * DIM / 256, 256, 0, stream>>>(W2, W2T);

    // embeddings
    embed_kernel<<<NTOK * 192 / 256, 256, 0, stream>>>(idx, emb, pos, xbf);

    // q, k, v projections (v stored transposed per batch)
    dim3 gg(NTOK / 128, DIM / 128);
    gemm_wmma_kernel<<<gg, 128, 0, stream>>>(xbf, WqT, nullptr, qbf, 0);
    gemm_wmma_kernel<<<gg, 128, 0, stream>>>(xbf, WkT, nullptr, kbf, 0);
    gemm_wmma_kernel<<<gg, 128, 0, stream>>>(xbf, WvT, nullptr, vTb, 1);

    // masked softmax attention
    dim3 ga(SEQ / 64, BATCH);
    attention_kernel<<<ga, 128, 4 * ATT_SLICE, stream>>>(qbf, kbf, vTb, amask, af32);

    // LN1 -> bf16
    layernorm_kernel<<<NTOK, 256, 0, stream>>>(af32, ln1g, ln1b, ln1o, 0);

    // FFN: gelu(x@W1+b1) -> bf16 ; gelu(h@W2+b2) -> f32 (reuse af32)
    gemm_wmma_kernel<<<gg, 128, 0, stream>>>(ln1o, W1T, b1, h1o, 2);
    gemm_wmma_kernel<<<gg, 128, 0, stream>>>(h1o, W2T, b2, af32, 3);

    // LN2 -> final f32 output
    layernorm_kernel<<<NTOK, 256, 0, stream>>>(af32, ln2g, ln2b, d_out, 1);
}
